// MultiheadAttention_61555471286490
// MI455X (gfx1250) — matrix operand order repaired and checked by reading, not compiled, not run on hardware
//
#include <hip/hip_runtime.h>
#include <hip/hip_bf16.h>

// Problem constants (match reference)
#define BB 2
#define SS 2048
#define DD 1024
#define HH 16
#define HD 64

typedef __attribute__((ext_vector_type(16))) __bf16 bf16x16;
typedef __attribute__((ext_vector_type(8)))  float  f32x8;

union BFU { unsigned short u; __bf16 b; };

__device__ __forceinline__ unsigned short f2bf(float f) {
  unsigned u = __builtin_bit_cast(unsigned, f);
  u += 0x7FFFu + ((u >> 16) & 1u);          // round-to-nearest-even
  return (unsigned short)(u >> 16);
}

__device__ __forceinline__ f32x8 wmma_bf16(bf16x16 a, bf16x16 b, f32x8 c) {
  // D = A(16x32 bf16) * B(32x16 bf16) + C(16x16 f32)
  return __builtin_amdgcn_wmma_f32_16x16x32_bf16(false, a, false, b, (short)0, c,
                                                 false, false);
}

// gfx1250 async DMA: global -> LDS, 16 bytes per enabled lane, ASYNCcnt.
// LDS destination offset = low 32 bits of the generic shared address
// (see ISA 10.2, LDS aperture).
__device__ __forceinline__ void async_copy_b128(unsigned lds_byte_off,
                                                const void* gaddr) {
  asm volatile("global_load_async_to_lds_b128 %0, %1, off"
               :: "v"(lds_byte_off), "v"((unsigned long long)gaddr)
               : "memory");
}
__device__ __forceinline__ void wait_async() {
  asm volatile("s_wait_asynccnt 0x0" ::: "memory");
}

// A fragment 16(M)x32(K) from LDS row-major [m][k] (stride in shorts).
// Layout: lane L -> m=L%16, K range = (L/16)*16 .. +15 (matches B fragments).
__device__ __forceinline__ bf16x16 frag_a(const unsigned short* lds, int row0,
                                          int koff, int stride) {
  const int lane = threadIdx.x & 31;
  const unsigned short* p = lds + (size_t)(row0 + (lane & 15)) * stride + koff;
  const int kb = (lane >> 4) * 16;
  bf16x16 a;
#pragma unroll
  for (int i = 0; i < 8; ++i) { BFU t; t.u = p[kb + i];      a[i]     = t.b; }
#pragma unroll
  for (int i = 0; i < 8; ++i) { BFU t; t.u = p[kb + 8 + i];  a[i + 8] = t.b; }
  return a;
}

// B fragment 32(K)x16(N) from LDS stored K-major [k][n].
// lane L -> n=L%16, K range = (L/16)*16 .. +15.
__device__ __forceinline__ bf16x16 frag_b_kmaj(const unsigned short* lds, int n0,
                                               int koff, int stride) {
  const int lane = threadIdx.x & 31;
  const int n = n0 + (lane & 15);
  const int kb = koff + (lane >> 4) * 16;
  bf16x16 b;
#pragma unroll
  for (int i = 0; i < 16; ++i) { BFU t; t.u = lds[(size_t)(kb + i) * stride + n]; b[i] = t.b; }
  return b;
}

// B fragment when LDS holds B transposed, i.e. [n][k] row-major (K contiguous).
__device__ __forceinline__ bf16x16 frag_b_nmaj(const unsigned short* lds, int n0,
                                               int koff, int stride) {
  const int lane = threadIdx.x & 31;
  const unsigned short* p = lds + (size_t)(n0 + (lane & 15)) * stride + koff +
                            (lane >> 4) * 16;
  bf16x16 b;
#pragma unroll
  for (int i = 0; i < 16; ++i) { BFU t; t.u = p[i]; b[i] = t.b; }
  return b;
}

// ---------------------------------------------------------------------------
// Kernel 1: QKV projection.  [4096,1024] x [1024,1024] + bias, output stored
// bf16 in per-head layout [B,H,S,64] honoring the Fortran head split
// (head h <- channels h, h+16, ..., h+16*63).
// blockIdx.z selects q/k/v.  Block tile 128x128, K-chunk 32.
// ---------------------------------------------------------------------------
__global__ __launch_bounds__(256) void qkv_proj_kernel(
    const float* __restrict__ qin, const float* __restrict__ kin,
    const float* __restrict__ vin, const float* __restrict__ Wq,
    const float* __restrict__ Wk, const float* __restrict__ Wv,
    const float* __restrict__ bq, const float* __restrict__ bk,
    const float* __restrict__ bv, unsigned short* __restrict__ qh,
    unsigned short* __restrict__ kh, unsigned short* __restrict__ vh) {
  const int which = blockIdx.z;
  const float* X = (which == 0) ? qin : (which == 1) ? kin : vin;
  const float* W = (which == 0) ? Wq : (which == 1) ? Wk : Wv;
  const float* Bv = (which == 0) ? bq : (which == 1) ? bk : bv;
  unsigned short* OUT = (which == 0) ? qh : (which == 1) ? kh : vh;

  __shared__ unsigned short ldsA[128 * 40];   // 128 x 32 bf16 (+pad)
  __shared__ unsigned short ldsB[32 * 136];   // 32 x 128 bf16 (+pad)

  const int t = threadIdx.x;
  const int lane = t & 31, wave = t >> 5;
  const int mBase = blockIdx.x * 128;
  const int nBase = blockIdx.y * 128;
  const int m0 = (wave >> 2) * 64;            // wave grid 2(M) x 4(N)
  const int n0 = (wave & 3) * 32;

  f32x8 acc[4][2] = {};

  for (int kt = 0; kt < DD / 32; ++kt) {
    const int kBase = kt * 32;
    // Prefetch next K-tile into L2/WGP$ while this one is consumed.
    if (kt + 1 < DD / 32) {
      __builtin_prefetch(X + (size_t)(mBase + (t >> 1)) * DD + kBase + 32, 0, 0);
      __builtin_prefetch(W + (size_t)(kBase + 32 + (t >> 3)) * DD + nBase, 0, 0);
    }
    // A tile: 128x32 fp32 -> bf16
#pragma unroll
    for (int i = 0; i < 4; ++i) {
      int idx = t + 256 * i;                  // 1024 float4
      int r = idx >> 3, cv = (idx & 7) * 4;
      float4 f = *(const float4*)(X + (size_t)(mBase + r) * DD + kBase + cv);
      unsigned short* dst = &ldsA[r * 40 + cv];
      *(unsigned*)(dst)     = f2bf(f.x) | ((unsigned)f2bf(f.y) << 16);
      *(unsigned*)(dst + 2) = f2bf(f.z) | ((unsigned)f2bf(f.w) << 16);
    }
    // B tile: 32x128 fp32 -> bf16 (W is [K][N] row-major)
#pragma unroll
    for (int i = 0; i < 4; ++i) {
      int idx = t + 256 * i;
      int r = idx >> 5, cv = (idx & 31) * 4;
      float4 f = *(const float4*)(W + (size_t)(kBase + r) * DD + nBase + cv);
      unsigned short* dst = &ldsB[r * 136 + cv];
      *(unsigned*)(dst)     = f2bf(f.x) | ((unsigned)f2bf(f.y) << 16);
      *(unsigned*)(dst + 2) = f2bf(f.z) | ((unsigned)f2bf(f.w) << 16);
    }
    __syncthreads();

    bf16x16 bf0 = frag_b_kmaj(ldsB, n0, 0, 136);
    bf16x16 bf1 = frag_b_kmaj(ldsB, n0 + 16, 0, 136);
#pragma unroll
    for (int mt = 0; mt < 4; ++mt) {
      bf16x16 af = frag_a(ldsA, m0 + mt * 16, 0, 40);
      acc[mt][0] = wmma_bf16(af, bf0, acc[mt][0]);
      acc[mt][1] = wmma_bf16(af, bf1, acc[mt][1]);
    }
    __syncthreads();
  }

  // Epilogue: bias add, Fortran head-split scatter to [B,H,S,64] bf16
#pragma unroll
  for (int mt = 0; mt < 4; ++mt) {
#pragma unroll
    for (int nt = 0; nt < 2; ++nt) {
      int n = nBase + n0 + nt * 16 + (lane & 15);
      float bias = Bv[n];
      int h = n & 15, j = n >> 4;
#pragma unroll
      for (int e = 0; e < 8; ++e) {
        int m = mBase + m0 + mt * 16 + (lane >> 4) * 8 + e;
        int b = m >> 11, s = m & (SS - 1);
        OUT[(((size_t)(b * HH + h) * SS) + s) * HD + j] =
            f2bf(acc[mt][nt][e] + bias);
      }
    }
  }
}

// ---------------------------------------------------------------------------
// Kernel 2: scores = qh @ kh^T * 1/8 per (b,h).  Causal block skip.
// Tile 128(M) x 128(N), K = 64.  Tiles are pure bf16 copies: use async
// global->LDS DMA (ASYNCcnt) instead of the VGPR path.
// ---------------------------------------------------------------------------
__global__ __launch_bounds__(256) void scores_kernel(
    const unsigned short* __restrict__ qh, const unsigned short* __restrict__ kh,
    float* __restrict__ attn) {
  if (blockIdx.x > blockIdx.y) return;        // fully masked block
  const int z = blockIdx.z;                   // b*H + h
  const int mBase = blockIdx.y * 128, nBase = blockIdx.x * 128;

  __shared__ unsigned short ldsQ[128 * 80];
  __shared__ unsigned short ldsK[128 * 80];

  const unsigned short* qp = qh + (size_t)z * SS * HD;
  const unsigned short* kp = kh + (size_t)z * SS * HD;
  const int t = threadIdx.x, lane = t & 31, wave = t >> 5;

#pragma unroll
  for (int i = 0; i < 4; ++i) {               // 1024 x 16B per tile
    int idx = t + 256 * i;
    int r = idx >> 3, cv = (idx & 7) * 8;
    async_copy_b128((unsigned)(size_t)&ldsQ[r * 80 + cv],
                    qp + (size_t)(mBase + r) * HD + cv);
    async_copy_b128((unsigned)(size_t)&ldsK[r * 80 + cv],
                    kp + (size_t)(nBase + r) * HD + cv);
  }
  wait_async();
  __syncthreads();

  const int m0 = (wave >> 2) * 64, n0 = (wave & 3) * 32;
  f32x8 acc[4][2] = {};
#pragma unroll
  for (int kc = 0; kc < 2; ++kc) {
    bf16x16 bf0 = frag_b_nmaj(ldsK, n0, kc * 32, 80);       // B[k][n]=kh[n][k]
    bf16x16 bf1 = frag_b_nmaj(ldsK, n0 + 16, kc * 32, 80);
#pragma unroll
    for (int mt = 0; mt < 4; ++mt) {
      bf16x16 af = frag_a(ldsQ, m0 + mt * 16, kc * 32, 80);
      acc[mt][0] = wmma_bf16(af, bf0, acc[mt][0]);
      acc[mt][1] = wmma_bf16(af, bf1, acc[mt][1]);
    }
  }

  float* out = attn + (size_t)z * SS * SS;
#pragma unroll
  for (int mt = 0; mt < 4; ++mt) {
#pragma unroll
    for (int nt = 0; nt < 2; ++nt) {
#pragma unroll
      for (int e = 0; e < 8; ++e) {
        int m = mBase + m0 + mt * 16 + (lane >> 4) * 8 + e;
        int n = nBase + n0 + nt * 16 + (lane & 15);
        out[(size_t)m * SS + n] = acc[mt][nt][e] * 0.125f;  // 1/sqrt(64)
      }
    }
  }
}

// ---------------------------------------------------------------------------
// Kernel 3: causal softmax, in place over attn.  One wave per row; row cached
// in 64 regs/lane; masked tail written as exact zeros.
// ---------------------------------------------------------------------------
__global__ __launch_bounds__(256) void softmax_kernel(float* __restrict__ attn) {
  const int wave = threadIdx.x >> 5, lane = threadIdx.x & 31;
  const size_t row = (size_t)blockIdx.x * 8 + wave;   // 0 .. B*H*S-1
  const int s = (int)(row & (SS - 1));                // query position
  float* p = attn + row * SS;
  const int nIt = (s >> 5) + 1;                       // 32-wide chunks covering 0..s

  float vals[64];
  float mx = -3.0e38f;
#pragma unroll
  for (int it = 0; it < 64; ++it) {
    if (it < nIt) {
      int col = it * 32 + lane;
      float v = (col <= s) ? p[col] : -3.0e38f;
      vals[it] = v;
      mx = fmaxf(mx, v);
    }
  }
#pragma unroll
  for (int off = 16; off > 0; off >>= 1) mx = fmaxf(mx, __shfl_xor(mx, off, 32));

  float sum = 0.f;
#pragma unroll
  for (int it = 0; it < 64; ++it) {
    if (it < nIt) {
      int col = it * 32 + lane;
      float e = (col <= s) ? __expf(vals[it] - mx) : 0.f;
      vals[it] = e;
      sum += e;
    }
  }
#pragma unroll
  for (int off = 16; off > 0; off >>= 1) sum += __shfl_xor(sum, off, 32);
  const float inv = 1.0f / sum;

#pragma unroll
  for (int it = 0; it < 64; ++it) {
    int col = it * 32 + lane;
    p[col] = (it < nIt) ? vals[it] * inv : 0.f;       // masked -> exact 0
  }
}

// ---------------------------------------------------------------------------
// Kernel 4: values = attn @ vh per (b,h).  M=2048, N=64, K only over the
// unmasked (lower-triangular) blocks.  V tile staged via async DMA.
// Output scattered C-order: d = h*64+j.
// ---------------------------------------------------------------------------
__global__ __launch_bounds__(256) void pv_kernel(
    const float* __restrict__ attn, const unsigned short* __restrict__ vh,
    unsigned short* __restrict__ vals) {
  const int z = blockIdx.z;
  const int mBase = blockIdx.x * 128;

  __shared__ unsigned short ldsP[128 * 80];   // 128 x 64 bf16 (+pad)
  __shared__ unsigned short ldsV[64 * 80];    // 64 x 64 bf16 (+pad)

  const float* P = attn + (size_t)z * SS * SS;
  const unsigned short* Vp = vh + (size_t)z * SS * HD;
  const int t = threadIdx.x, lane = t & 31, wave = t >> 5;
  const int m0 = (wave >> 1) * 32, n0 = (wave & 1) * 32;  // wave grid 4(M) x 2(N)

  f32x8 acc[2][2] = {};
  const int numK = blockIdx.x * 2 + 2;        // K-blocks up to the diagonal

  for (int kt = 0; kt < numK; ++kt) {
    const int kBase = kt * 64;
#pragma unroll
    for (int i = 0; i < 2; ++i) {             // V tile: 512 x 16B, async DMA
      int idx = t + 256 * i;
      int r = idx >> 3, cv = (idx & 7) * 8;
      async_copy_b128((unsigned)(size_t)&ldsV[r * 80 + cv],
                      Vp + (size_t)(kBase + r) * HD + cv);
    }
#pragma unroll
    for (int i = 0; i < 8; ++i) {             // 2048 float4, attn fp32 -> bf16
      int idx = t + 256 * i;
      int r = idx >> 4, cv = (idx & 15) * 4;
      float4 f = *(const float4*)(P + (size_t)(mBase + r) * SS + kBase + cv);
      unsigned short* dst = &ldsP[r * 80 + cv];
      *(unsigned*)(dst)     = f2bf(f.x) | ((unsigned)f2bf(f.y) << 16);
      *(unsigned*)(dst + 2) = f2bf(f.z) | ((unsigned)f2bf(f.w) << 16);
    }
    wait_async();
    __syncthreads();

#pragma unroll
    for (int kc = 0; kc < 2; ++kc) {
      bf16x16 bf0 = frag_b_kmaj(ldsV, n0, kc * 32, 80);
      bf16x16 bf1 = frag_b_kmaj(ldsV, n0 + 16, kc * 32, 80);
#pragma unroll
      for (int mt = 0; mt < 2; ++mt) {
        bf16x16 af = frag_a(ldsP, m0 + mt * 16, kc * 32, 80);
        acc[mt][0] = wmma_bf16(af, bf0, acc[mt][0]);
        acc[mt][1] = wmma_bf16(af, bf1, acc[mt][1]);
      }
    }
    __syncthreads();
  }

  const int b = z >> 4, h = z & 15;
#pragma unroll
  for (int mt = 0; mt < 2; ++mt) {
#pragma unroll
    for (int nt = 0; nt < 2; ++nt) {
#pragma unroll
      for (int e = 0; e < 8; ++e) {
        int s = mBase + m0 + mt * 16 + (lane >> 4) * 8 + e;
        int j = n0 + nt * 16 + (lane & 15);
        vals[((size_t)(b * SS + s)) * DD + h * HD + j] = f2bf(acc[mt][nt][e]);
      }
    }
  }
}

// ---------------------------------------------------------------------------
// Kernel 5: out = values @ Wo + bo (fp32 out).  A is already bf16 in
// workspace -> async DMA staging; Wo converts fp32->bf16 on the VGPR path.
// ---------------------------------------------------------------------------
__global__ __launch_bounds__(256) void out_proj_kernel(
    const unsigned short* __restrict__ vals, const float* __restrict__ Wo,
    const float* __restrict__ bo, float* __restrict__ out) {
  __shared__ unsigned short ldsA[128 * 40];
  __shared__ unsigned short ldsB[32 * 136];

  const int t = threadIdx.x, lane = t & 31, wave = t >> 5;
  const int mBase = blockIdx.x * 128, nBase = blockIdx.y * 128;
  const int m0 = (wave >> 2) * 64, n0 = (wave & 3) * 32;

  f32x8 acc[4][2] = {};

  for (int kt = 0; kt < DD / 32; ++kt) {
    const int kBase = kt * 32;
    if (kt + 1 < DD / 32) {
      __builtin_prefetch(Wo + (size_t)(kBase + 32 + (t >> 3)) * DD + nBase, 0, 0);
    }
#pragma unroll
    for (int i = 0; i < 2; ++i) {             // A tile: 512 x 16B, async DMA
      int idx = t + 256 * i;
      int r = idx >> 2, cv = (idx & 3) * 8;
      async_copy_b128((unsigned)(size_t)&ldsA[r * 40 + cv],
                      vals + (size_t)(mBase + r) * DD + kBase + cv);
    }
#pragma unroll
    for (int i = 0; i < 4; ++i) {             // Wo fp32 -> bf16
      int idx = t + 256 * i;
      int r = idx >> 5, cv = (idx & 31) * 4;
      float4 f = *(const float4*)(Wo + (size_t)(kBase + r) * DD + nBase + cv);
      unsigned short* dst = &ldsB[r * 136 + cv];
      *(unsigned*)(dst)     = f2bf(f.x) | ((unsigned)f2bf(f.y) << 16);
      *(unsigned*)(dst + 2) = f2bf(f.z) | ((unsigned)f2bf(f.w) << 16);
    }
    wait_async();
    __syncthreads();

    bf16x16 bf0 = frag_b_kmaj(ldsB, n0, 0, 136);
    bf16x16 bf1 = frag_b_kmaj(ldsB, n0 + 16, 0, 136);
#pragma unroll
    for (int mt = 0; mt < 4; ++mt) {
      bf16x16 af = frag_a(ldsA, m0 + mt * 16, 0, 40);
      acc[mt][0] = wmma_bf16(af, bf0, acc[mt][0]);
      acc[mt][1] = wmma_bf16(af, bf1, acc[mt][1]);
    }
    __syncthreads();
  }

#pragma unroll
  for (int mt = 0; mt < 4; ++mt) {
#pragma unroll
    for (int nt = 0; nt < 2; ++nt) {
      int n = nBase + n0 + nt * 16 + (lane & 15);
      float bias = bo[n];
#pragma unroll
      for (int e = 0; e < 8; ++e) {
        int m = mBase + m0 + mt * 16 + (lane >> 4) * 8 + e;
        out[(size_t)m * DD + n] = acc[mt][nt][e] + bias;
      }
    }
  }
}

// ---------------------------------------------------------------------------
extern "C" void kernel_launch(void* const* d_in, const int* in_sizes, int n_in,
                              void* d_out, int out_size, void* d_ws,
                              size_t ws_size, hipStream_t stream) {
  const float* q  = (const float*)d_in[0];
  const float* k  = (const float*)d_in[1];
  const float* v  = (const float*)d_in[2];
  // d_in[3] = mask: causal tril, handled analytically in-kernel
  const float* Wq = (const float*)d_in[4];
  const float* bq = (const float*)d_in[5];
  const float* Wk = (const float*)d_in[6];
  const float* bk = (const float*)d_in[7];
  const float* Wv = (const float*)d_in[8];
  const float* bv = (const float*)d_in[9];
  const float* Wo = (const float*)d_in[10];
  const float* bo = (const float*)d_in[11];

  float* out  = (float*)d_out;                       // [B,S,D]
  float* attn = out + (size_t)BB * SS * DD;          // [B,H,S,S]

  const size_t headElems = (size_t)BB * HH * SS * HD;  // 4M
  unsigned short* qh   = (unsigned short*)d_ws;
  unsigned short* kh   = qh + headElems;
  unsigned short* vh   = kh + headElems;
  unsigned short* vals = vh + headElems;             // [B,S,D] bf16

  qkv_proj_kernel<<<dim3(32, 8, 3), 256, 0, stream>>>(q, k, v, Wq, Wk, Wv, bq,
                                                      bk, bv, qh, kh, vh);
  scores_kernel<<<dim3(16, 16, BB * HH), 256, 0, stream>>>(qh, kh, attn);
  softmax_kernel<<<dim3((BB * HH * SS) / 8), 256, 0, stream>>>(attn);
  pv_kernel<<<dim3(16, 1, BB * HH), 256, 0, stream>>>(attn, vh, vals);
  out_proj_kernel<<<dim3(32, 8, 1), 256, 0, stream>>>(vals, Wo, bo, out);
}
